// MusicAttrVAE_38139309588578
// MI455X (gfx1250) — matrix operand order repaired and checked by reading, not compiled, not run on hardware
//
#include <hip/hip_runtime.h>
#include <hip/hip_bf16.h>
#include <cstdint>
#include <cstddef>

#define B_    256
#define S_    256
#define ROLL_ 130
#define RHY_  3
#define NOTE_ 16
#define CHR_  24
#define H_    512
#define Z_    128
#define G3_   1536   /* 3*H */

typedef __attribute__((ext_vector_type(16))) __bf16 v16bf;
typedef __attribute__((ext_vector_type(8)))  float  v8f;

union Frag { v16bf v; uint4 q[2]; };

__device__ __forceinline__ unsigned short f2bf(float x) {
  unsigned u = __float_as_uint(x);
  u += 0x7FFFu + ((u >> 16) & 1u);
  return (unsigned short)(u >> 16);
}
__device__ __forceinline__ float sigf(float x) { return 1.0f / (1.0f + expf(-x)); }

// ---- CDNA5 async global->LDS staging (guarded; falls back to ld+ds_store) --
#if defined(__AMDGCN__) && __has_builtin(__builtin_amdgcn_global_load_async_to_lds_b128)
#define HAVE_ASYNC_LDS 1
#endif

typedef int asi4 __attribute__((vector_size(16)));              // <4 x i32>
typedef __attribute__((address_space(1))) asi4* gld_v4i;        // global int4*
typedef __attribute__((address_space(3))) asi4* lds_v4i;        // LDS int4*

__device__ __forceinline__ void stage16(const unsigned short* g,
                                        unsigned short* lds) {
#if defined(HAVE_ASYNC_LDS)
  __builtin_amdgcn_global_load_async_to_lds_b128((gld_v4i)g, (lds_v4i)lds, 0, 0);
#else
  *(uint4*)lds = *(const uint4*)g;
#endif
}

__device__ __forceinline__ void stage_wait() {
#if defined(HAVE_ASYNC_LDS)
#if __has_builtin(__builtin_amdgcn_s_wait_asynccnt)
  __builtin_amdgcn_s_wait_asynccnt(0);
#else
  asm volatile("s_wait_asynccnt 0x0" ::: "memory");
#endif
#endif
}

// ---------------------------------------------------------------------------
// C[M,N] = A[M,K](bf16) * W[N,K]^T(bf16) + bias[N], f32 out.
// 64x64 tile, 8 waves, 2 WMMA tiles/wave. For small/medium GEMMs (keeps grid
// wide for the latency-bound per-timestep recurrent GEMMs).
// ---------------------------------------------------------------------------
__global__ __launch_bounds__(256) void k_gemm_bf16(
    const unsigned short* __restrict__ A,
    const unsigned short* __restrict__ W,
    const float* __restrict__ bias,
    float* __restrict__ C,
    int M, int N, int K)
{
  __shared__ __align__(16) unsigned short As[64 * 32];
  __shared__ __align__(16) unsigned short Ws[64 * 32];
  const int n0 = blockIdx.x * 64;
  const int m0 = blockIdx.y * 64;
  const int t = threadIdx.x;
  const int lane = t & 31, wave = t >> 5;
  const int mt = wave >> 1;            // 0..3  (16-row tile)
  const int nt0 = (wave & 1) * 2;      // 0 or 2 (two 16-col tiles)
  const int half = lane >> 4, lm = lane & 15;
  const int srow = t >> 2, schunk = (t & 3) * 8;

  v8f acc0 = {};
  v8f acc1 = {};

  for (int k0 = 0; k0 < K; k0 += 32) {
    stage16(A + (size_t)(m0 + srow) * K + k0 + schunk, &As[srow * 32 + schunk]);
    stage16(W + (size_t)(n0 + srow) * K + k0 + schunk, &Ws[srow * 32 + schunk]);
    if (k0 + 32 < K) {
      __builtin_prefetch(A + (size_t)(m0 + srow) * K + k0 + 32 + schunk, 0, 0);
      __builtin_prefetch(W + (size_t)(n0 + srow) * K + k0 + 32 + schunk, 0, 0);
    }
    stage_wait();
    __syncthreads();

    // A fragment (16x32 bf16): half0 -> K0..15; half1 -> K16..31
    Frag a, b0, b1;
    const unsigned short* ap = &As[(mt * 16 + lm) * 32 + half * 16];
    a.q[0] = *(const uint4*)(ap);
    a.q[1] = *(const uint4*)(ap + 8);
    // B fragment (32x16 bf16): lane holds col n=lm; half0 K0..15, half1 K16..31
    const unsigned short* bp0 = &Ws[((nt0 + 0) * 16 + lm) * 32 + half * 16];
    b0.q[0] = *(const uint4*)(bp0);
    b0.q[1] = *(const uint4*)(bp0 + 8);
    const unsigned short* bp1 = &Ws[((nt0 + 1) * 16 + lm) * 32 + half * 16];
    b1.q[0] = *(const uint4*)(bp1);
    b1.q[1] = *(const uint4*)(bp1 + 8);

    acc0 = __builtin_amdgcn_wmma_f32_16x16x32_bf16(false, a.v, false, b0.v,
                                                   (short)0, acc0, false, false);
    acc1 = __builtin_amdgcn_wmma_f32_16x16x32_bf16(false, a.v, false, b1.v,
                                                   (short)0, acc1, false, false);
    __syncthreads();
  }

  for (int p2 = 0; p2 < 2; ++p2) {
    v8f acc = p2 ? acc1 : acc0;
    int nt = nt0 + p2;
    int gn = n0 + nt * 16 + lm;
    float bv = bias ? bias[gn] : 0.0f;
    for (int v = 0; v < 8; ++v) {
      int gm = m0 + mt * 16 + v + half * 8;
      C[(size_t)gm * N + gn] = acc[v] + bv;
    }
  }
}

// ---------------------------------------------------------------------------
// 128x128 tile, 8 waves, each wave 2x4 = 8 WMMA tiles, double-buffered LDS
// (one barrier per 32-wide K slab). Used for the big M>=512 GEMMs.
// ---------------------------------------------------------------------------
__global__ __launch_bounds__(256) void k_gemm128_bf16(
    const unsigned short* __restrict__ A,
    const unsigned short* __restrict__ W,
    const float* __restrict__ bias,
    float* __restrict__ C,
    int M, int N, int K)
{
  __shared__ __align__(16) unsigned short As[2][128 * 32];
  __shared__ __align__(16) unsigned short Ws[2][128 * 32];
  const int n0 = blockIdx.x * 128;
  const int m0 = blockIdx.y * 128;
  const int t = threadIdx.x;
  const int lane = t & 31, wave = t >> 5;
  const int mtb = (wave >> 1) * 2;     // m-tile base: 0,2,4,6
  const int ntb = (wave & 1) * 4;      // n-tile base: 0 or 4
  const int half = lane >> 4, lm = lane & 15;

  v8f acc[2][4];
  {
    v8f zz = {};
    for (int i = 0; i < 2; ++i)
      for (int j = 0; j < 4; ++j) acc[i][j] = zz;
  }

  // staging: 128 rows x 32 cols = 512 16B chunks; 2 chunks/thread per matrix
  const int c0 = t * 2, c1 = t * 2 + 1;
  const int ar0 = c0 >> 2, ao0 = (c0 & 3) * 8;
  const int ar1 = c1 >> 2, ao1 = (c1 & 3) * 8;

  auto stage = [&](int buf, int k0) {
    stage16(A + (size_t)(m0 + ar0) * K + k0 + ao0, &As[buf][ar0 * 32 + ao0]);
    stage16(A + (size_t)(m0 + ar1) * K + k0 + ao1, &As[buf][ar1 * 32 + ao1]);
    stage16(W + (size_t)(n0 + ar0) * K + k0 + ao0, &Ws[buf][ar0 * 32 + ao0]);
    stage16(W + (size_t)(n0 + ar1) * K + k0 + ao1, &Ws[buf][ar1 * 32 + ao1]);
  };
  auto compute = [&](int buf) {
    Frag a[2], b[4];
    for (int i = 0; i < 2; ++i) {
      const unsigned short* ap = &As[buf][((mtb + i) * 16 + lm) * 32 + half * 16];
      a[i].q[0] = *(const uint4*)(ap);
      a[i].q[1] = *(const uint4*)(ap + 8);
    }
    for (int j = 0; j < 4; ++j) {
      const unsigned short* bp = &Ws[buf][((ntb + j) * 16 + lm) * 32 + half * 16];
      b[j].q[0] = *(const uint4*)(bp);
      b[j].q[1] = *(const uint4*)(bp + 8);
    }
    for (int i = 0; i < 2; ++i)
      for (int j = 0; j < 4; ++j)
        acc[i][j] = __builtin_amdgcn_wmma_f32_16x16x32_bf16(
            false, a[i].v, false, b[j].v, (short)0, acc[i][j], false, false);
  };

  stage(0, 0);
  stage_wait();
  __syncthreads();
  int cur = 0;
  for (int k0 = 32; k0 < K; k0 += 32) {
    stage(cur ^ 1, k0);   // fill next buffer while computing on current
    compute(cur);
    stage_wait();
    __syncthreads();
    cur ^= 1;
  }
  compute(cur);

  for (int i = 0; i < 2; ++i) {
    for (int j = 0; j < 4; ++j) {
      int gn = n0 + (ntb + j) * 16 + lm;
      float bv = bias ? bias[gn] : 0.0f;
      for (int v = 0; v < 8; ++v) {
        int gm = m0 + (mtb + i) * 16 + v + half * 8;
        C[(size_t)gm * N + gn] = acc[i][j][v] + bv;
      }
    }
  }
}

// ------------------------- elementwise / pack kernels ----------------------
__global__ void k_pack_w(size_t n, const float* __restrict__ src,
                         unsigned short* __restrict__ dst,
                         int N, int K, int NP, int KP) {
  size_t i = (size_t)blockIdx.x * 256 + threadIdx.x;
  if (i >= n) return;
  int r = (int)(i / KP), k = (int)(i % KP);
  float v = (r < N && k < K) ? src[(size_t)r * K + k] : 0.0f;
  dst[i] = f2bf(v);
}

__global__ void k_pack_bias(size_t n, const float* __restrict__ src,
                            float* __restrict__ dst, int N, int NP) {
  size_t i = (size_t)blockIdx.x * 256 + threadIdx.x;
  if (i >= n) return;
  (void)NP;
  dst[i] = ((int)i < N) ? src[i] : 0.0f;
}

__global__ void k_cvt_bf16(size_t n, const float* __restrict__ src,
                           unsigned short* __restrict__ dst) {
  size_t i = (size_t)blockIdx.x * 256 + threadIdx.x;
  if (i >= n) return;
  dst[i] = f2bf(src[i]);
}

__global__ void k_zero_f(size_t n, float* p) {
  size_t i = (size_t)blockIdx.x * 256 + threadIdx.x;
  if (i < n) p[i] = 0.0f;
}
__global__ void k_zero_h(size_t n, unsigned short* p) {
  size_t i = (size_t)blockIdx.x * 256 + threadIdx.x;
  if (i < n) p[i] = 0;
}

// dst rows ordered (s major, b minor): row = s*B + b; cols [x(b,ss,:Dx)|cond(b,:Dc)|0]
__global__ void k_build_seq(size_t n, const float* __restrict__ x,
                            const float* __restrict__ cond,
                            unsigned short* __restrict__ dst,
                            int B, int S, int Dx, int Dc, int KP, int rev) {
  size_t i = (size_t)blockIdx.x * 256 + threadIdx.x;
  if (i >= n) return;
  int k = (int)(i % KP);
  size_t row = i / KP;
  int b = (int)(row % B);
  int s = (int)(row / B);
  int ss = rev ? (S - 1 - s) : s;
  float v = 0.0f;
  if (k < Dx)            v = x[((size_t)b * S + ss) * Dx + k];
  else if (k < Dx + Dc)  v = cond[(size_t)b * Dc + (k - Dx)];
  dst[i] = f2bf(v);
}

__global__ void k_build_z(size_t n, const float* zr, const float* cr,
                          const float* zn, const float* cn,
                          const float* zc, const float* ch, float* dst) {
  size_t i = (size_t)blockIdx.x * 256 + threadIdx.x;
  if (i >= n) return;
  int b = (int)(i / 414), c = (int)(i % 414);
  float v;
  if (c < 128)       v = zr[b * 128 + c];
  else if (c < 131)  v = cr[b * 3 + (c - 128)];
  else if (c < 259)  v = zn[b * 128 + (c - 131)];
  else if (c < 262)  v = cn[b * 3 + (c - 259)];
  else if (c < 390)  v = zc[b * 128 + (c - 262)];
  else               v = ch[b * 24 + (c - 390)];
  dst[i] = v;
}

__global__ void k_build_g_inp(size_t n, const float* __restrict__ onehot,
                              const float* __restrict__ z,
                              unsigned short* __restrict__ dst) {
  size_t i = (size_t)blockIdx.x * 256 + threadIdx.x;
  if (i >= n) return;
  int b = (int)(i / 544), k = (int)(i % 544);
  float v = (k < ROLL_) ? onehot[b * ROLL_ + k] : z[b * 414 + (k - ROLL_)];
  dst[i] = f2bf(v);
}

__global__ void k_init_onehot(size_t n, float* dst) {
  size_t i = (size_t)blockIdx.x * 256 + threadIdx.x;
  if (i >= n) return;
  dst[i] = ((int)(i % ROLL_) == ROLL_ - 1) ? 1.0f : 0.0f;
}

__global__ void k_copy_cols(size_t n, const float* __restrict__ src,
                            float* __restrict__ dst, int Cs, int dstStride,
                            int colOff) {
  size_t i = (size_t)blockIdx.x * 256 + threadIdx.x;
  if (i >= n) return;
  int r = (int)(i / Cs), c = (int)(i % Cs);
  dst[(size_t)r * dstStride + colOff + c] = src[i];
}

// PyTorch GRU gate fusion (r,z,n). gi/gh each [B,3H]; in-place h update safe.
__global__ void k_gru_fuse(size_t n, const float* __restrict__ gi,
                           const float* __restrict__ gh,
                           const float* h_in, float* h_out,
                           unsigned short* __restrict__ h_out_bf,
                           float* __restrict__ seq, int B, int H) {
  size_t i = (size_t)blockIdx.x * 256 + threadIdx.x;
  if (i >= n) return;
  (void)B;
  int row = (int)(i / H), col = (int)(i % H);
  const float* gir = gi + (size_t)row * 3 * H;
  const float* ghr = gh + (size_t)row * 3 * H;
  float r  = sigf(gir[col] + ghr[col]);
  float zg = sigf(gir[H + col] + ghr[H + col]);
  float nn = tanhf(gir[2 * H + col] + r * ghr[2 * H + col]);
  float h  = (1.0f - zg) * nn + zg * h_in[i];
  h_out[i] = h;
  h_out_bf[i] = f2bf(h);
  if (seq) seq[i] = h;
}

__global__ void k_reparam(size_t n, const float* __restrict__ mu,
                          const float* __restrict__ varlin,
                          const float* __restrict__ eps,
                          float* __restrict__ z, unsigned short* __restrict__ zbf,
                          float* __restrict__ out_mu, float* __restrict__ out_var) {
  size_t i = (size_t)blockIdx.x * 256 + threadIdx.x;
  if (i >= n) return;
  float m = mu[i];
  float v = expf(varlin[i]);
  float zz = m + v * eps[i];
  out_mu[i] = m;
  out_var[i] = v;
  z[i] = zz;
  zbf[i] = f2bf(zz);
}

// log_softmax over the time axis; src rows ordered s*B+b, padded stride NP.
__global__ void k_logsoftmax_time(size_t n, const float* __restrict__ src,
                                  float* __restrict__ dst, int B, int S, int F,
                                  int NP) {
  size_t i = (size_t)blockIdx.x * 256 + threadIdx.x;
  if (i >= n) return;
  int b = (int)(i / F), j = (int)(i % F);
  float mx = -3.4e38f;
  for (int s = 0; s < S; ++s)
    mx = fmaxf(mx, src[((size_t)s * B + b) * NP + j]);
  float sum = 0.0f;
  for (int s = 0; s < S; ++s)
    sum += expf(src[((size_t)s * B + b) * NP + j] - mx);
  float lse = mx + logf(sum);
  for (int s = 0; s < S; ++s)
    dst[((size_t)b * S + s) * F + j] = src[((size_t)s * B + b) * NP + j] - lse;
}

// AR epilogue: per-row log_softmax over 130 logits -> out_g[b,t,:]; argmax->one-hot.
__global__ void k_lsm_argmax(size_t n, const float* __restrict__ logits,
                             float* __restrict__ outg, int t,
                             float* __restrict__ onehot) {
  size_t b = (size_t)blockIdx.x * 256 + threadIdx.x;
  if (b >= n) return;
  const float* row = logits + b * 192;
  float mx = -3.4e38f;
  int am = 0;
  for (int j = 0; j < ROLL_; ++j)
    if (row[j] > mx) { mx = row[j]; am = j; }
  float sum = 0.0f;
  for (int j = 0; j < ROLL_; ++j) sum += expf(row[j] - mx);
  float lse = mx + logf(sum);
  float* dst = outg + (b * S_ + t) * ROLL_;
  float* oh  = onehot + b * ROLL_;
  for (int j = 0; j < ROLL_; ++j) {
    dst[j] = row[j] - lse;
    oh[j]  = (j == am) ? 1.0f : 0.0f;
  }
}

__global__ void k_sigmoid_slice(size_t n, const float* __restrict__ src,
                                float* __restrict__ dst, int C, int NP) {
  size_t i = (size_t)blockIdx.x * 256 + threadIdx.x;
  if (i >= n) return;
  int b = (int)(i / C), j = (int)(i % C);
  dst[i] = sigf(src[(size_t)b * NP + j]);
}

// ---------------------------------------------------------------------------
extern "C" void kernel_launch(void* const* d_in, const int* in_sizes, int n_in,
                              void* d_out, int out_size, void* d_ws, size_t ws_size,
                              hipStream_t stream) {
  (void)in_sizes; (void)n_in; (void)out_size; (void)ws_size;
  auto F32 = [&](int i) { return (const float*)d_in[i]; };
  float* out = (float*)d_out;

  char* wsp = (char*)d_ws;
  auto alloc = [&](size_t bytes) -> char* {
    char* r = wsp;
    wsp += (bytes + 255) & ~((size_t)255);
    return r;
  };

  float* gi_big = (float*)alloc((size_t)S_ * B_ * G3_ * 4);
  unsigned short* seq_in = (unsigned short*)alloc((size_t)S_ * B_ * 160 * 2);
  float* seqh = (float*)alloc((size_t)S_ * B_ * H_ * 4);
  unsigned short* seqh_bf = (unsigned short*)alloc((size_t)S_ * B_ * H_ * 2);
  float* seq_out = (float*)alloc((size_t)S_ * B_ * 64 * 4);

  unsigned short* wihB = (unsigned short*)alloc((size_t)G3_ * 160 * 2);
  unsigned short* whhB = (unsigned short*)alloc((size_t)G3_ * H_ * 2);
  unsigned short* muB = (unsigned short*)alloc((size_t)Z_ * 1024 * 2);
  unsigned short* varB = (unsigned short*)alloc((size_t)Z_ * 1024 * 2);
  unsigned short* liB = (unsigned short*)alloc((size_t)H_ * Z_ * 2);
  unsigned short* ligB = (unsigned short*)alloc((size_t)H_ * 416 * 2);
  unsigned short* cgWihB = (unsigned short*)alloc((size_t)G3_ * 544 * 2);
  unsigned short* cgWhhB = (unsigned short*)alloc((size_t)G3_ * H_ * 2);
  unsigned short* cg2WihB = (unsigned short*)alloc((size_t)G3_ * H_ * 2);
  unsigned short* cg2WhhB = (unsigned short*)alloc((size_t)G3_ * H_ * 2);
  unsigned short* outWB = (unsigned short*)alloc((size_t)64 * H_ * 2);
  unsigned short* outcWB = (unsigned short*)alloc((size_t)64 * Z_ * 2);
  unsigned short* outgWB = (unsigned short*)alloc((size_t)192 * H_ * 2);
  float* outBias = (float*)alloc(192 * 4);
  float* outgBias = (float*)alloc(192 * 4);

  float* hEnc = (float*)alloc((size_t)B_ * H_ * 4);
  unsigned short* hEncB = (unsigned short*)alloc((size_t)B_ * H_ * 2);
  float* gh_buf = (float*)alloc((size_t)B_ * G3_ * 4);
  float* hcat = (float*)alloc((size_t)B_ * 1024 * 4);
  unsigned short* hcatB = (unsigned short*)alloc((size_t)B_ * 1024 * 2);
  float* muP = (float*)alloc((size_t)B_ * Z_ * 4);
  float* varP = (float*)alloc((size_t)B_ * Z_ * 4);
  float* zf[3];
  unsigned short* zb[3];
  for (int i = 0; i < 3; ++i) {
    zf[i] = (float*)alloc((size_t)B_ * Z_ * 4);
    zb[i] = (unsigned short*)alloc((size_t)B_ * Z_ * 2);
  }
  float* z414 = (float*)alloc((size_t)B_ * 414 * 4);
  unsigned short* z416B = (unsigned short*)alloc((size_t)B_ * 416 * 2);
  float* onehot = (float*)alloc((size_t)B_ * ROLL_ * 4);
  unsigned short* inpgB = (unsigned short*)alloc((size_t)B_ * 544 * 2);
  float* giG = (float*)alloc((size_t)B_ * G3_ * 4);
  float* hx0 = (float*)alloc((size_t)B_ * H_ * 4);
  unsigned short* hx0B = (unsigned short*)alloc((size_t)B_ * H_ * 2);
  float* hx1 = (float*)alloc((size_t)B_ * H_ * 4);
  unsigned short* hx1B = (unsigned short*)alloc((size_t)B_ * H_ * 2);
  float* logitsP = (float*)alloc((size_t)B_ * 192 * 4);
  float* ctmp = (float*)alloc((size_t)B_ * 64 * 4);

  auto L = [&](auto kf, size_t n, auto&&... args) {
    unsigned g = (unsigned)((n + 255) / 256);
    kf<<<dim3(g), dim3(256), 0, stream>>>(n, args...);
  };
  auto GEMM = [&](const unsigned short* A, const unsigned short* W,
                  const float* bias, float* C, int M, int N, int K) {
    if (M >= 512 && (M % 128) == 0 && (N % 128) == 0)
      k_gemm128_bf16<<<dim3(N / 128, M / 128), dim3(256), 0, stream>>>(
          A, W, bias, C, M, N, K);
    else
      k_gemm_bf16<<<dim3(N / 64, M / 64), dim3(256), 0, stream>>>(
          A, W, bias, C, M, N, K);
  };

  const size_t OFF_OUTG = 0;
  const size_t OFF_R = OFF_OUTG + (size_t)B_ * S_ * ROLL_;
  const size_t OFF_N = OFF_R + (size_t)B_ * S_ * RHY_;
  const size_t OFF_C = OFF_N + (size_t)B_ * S_ * NOTE_;
  const size_t OFF_MUR = OFF_C + (size_t)B_ * CHR_;  // mu_r,var_r,mu_n,var_n,mu_c,var_c

  const int P_ENC[6] = {9, 13, 17, 21, 25, 29};  // Wih; +1 Whh; +2 bih; +3 bhh
  const int P_MU[6] = {33, 35, 37, 39, 41, 43};  // W; +1 b
  const int P_LIG = 45, P_CG = 47, P_CG2 = 51, P_LIR = 55, P_LIN = 57;
  const int P_DR = 59, P_DN = 63;
  const int P_OUTR = 67, P_OUTN = 69, P_OUTC = 71, P_OUTG = 73;

  // ---------------- encoders: 3 bi-GRUs -> mu/var -> z ----------------
  const float* encCond[6] = {F32(4), F32(4), F32(5), F32(5), F32(3), F32(3)};
  const int encDc[6] = {3, 3, 3, 3, 24, 24};
  for (int e = 0; e < 6; ++e) {
    int pw = P_ENC[e];
    int Kact = ROLL_ + encDc[e];
    L(k_pack_w, (size_t)G3_ * 160, F32(pw), wihB, G3_, Kact, G3_, 160);
    L(k_pack_w, (size_t)G3_ * H_, F32(pw + 1), whhB, G3_, H_, G3_, H_);
    L(k_build_seq, (size_t)S_ * B_ * 160, F32(0), encCond[e], seq_in, B_, S_,
      ROLL_, encDc[e], 160, e & 1);
    GEMM(seq_in, wihB, F32(pw + 2), gi_big, S_ * B_, G3_, 160);
    L(k_zero_f, (size_t)B_ * H_, hEnc);
    L(k_zero_h, (size_t)B_ * H_, hEncB);
    for (int t = 0; t < S_; ++t) {
      GEMM(hEncB, whhB, F32(pw + 3), gh_buf, B_, G3_, H_);
      L(k_gru_fuse, (size_t)B_ * H_,
        (const float*)(gi_big + (size_t)t * B_ * G3_), (const float*)gh_buf,
        (const float*)hEnc, hEnc, hEncB, (float*)nullptr, B_, H_);
    }
    L(k_copy_cols, (size_t)B_ * H_, (const float*)hEnc, hcat, H_, 1024,
      (e & 1) * H_);
    if (e & 1) {
      int pi = e >> 1;  // 0=r, 1=n, 2=c
      L(k_cvt_bf16, (size_t)B_ * 1024, (const float*)hcat, hcatB);
      L(k_pack_w, (size_t)Z_ * 1024, F32(P_MU[2 * pi]), muB, Z_, 1024, Z_, 1024);
      L(k_pack_w, (size_t)Z_ * 1024, F32(P_MU[2 * pi + 1]), varB, Z_, 1024, Z_, 1024);
      GEMM(hcatB, muB, F32(P_MU[2 * pi] + 1), muP, B_, Z_, 1024);
      GEMM(hcatB, varB, F32(P_MU[2 * pi + 1] + 1), varP, B_, Z_, 1024);
      L(k_reparam, (size_t)B_ * Z_, (const float*)muP, (const float*)varP,
        F32(6 + pi), zf[pi], zb[pi],
        out + OFF_MUR + (size_t)(pi * 2) * B_ * Z_,
        out + OFF_MUR + (size_t)(pi * 2 + 1) * B_ * Z_);
    }
  }

  // ---------------- sub-decoders (rhythm, note) ----------------
  const float* decX[2] = {F32(1), F32(2)};
  const int decDx[2] = {RHY_, NOTE_};
  const int decPW[2] = {P_DR, P_DN};
  const int decPLI[2] = {P_LIR, P_LIN};
  const int decPOUT[2] = {P_OUTR, P_OUTN};
  const size_t decOff[2] = {OFF_R, OFF_N};
  for (int d = 0; d < 2; ++d) {
    int pw = decPW[d];
    int Kact = decDx[d] + Z_;
    L(k_pack_w, (size_t)G3_ * 160, F32(pw), wihB, G3_, Kact, G3_, 160);
    L(k_pack_w, (size_t)G3_ * H_, F32(pw + 1), whhB, G3_, H_, G3_, H_);
    L(k_pack_w, (size_t)H_ * Z_, F32(decPLI[d]), liB, H_, Z_, H_, Z_);
    L(k_pack_w, (size_t)64 * H_, F32(decPOUT[d]), outWB, decDx[d], H_, 64, H_);
    L(k_pack_bias, (size_t)64, F32(decPOUT[d] + 1), outBias, decDx[d], 64);
    GEMM(zb[d], liB, F32(decPLI[d] + 1), hEnc, B_, H_, Z_);
    L(k_cvt_bf16, (size_t)B_ * H_, (const float*)hEnc, hEncB);
    L(k_build_seq, (size_t)S_ * B_ * 160, decX[d], (const float*)zf[d], seq_in,
      B_, S_, decDx[d], Z_, 160, 0);
    GEMM(seq_in, wihB, F32(pw + 2), gi_big, S_ * B_, G3_, 160);
    for (int t = 0; t < S_; ++t) {
      GEMM(hEncB, whhB, F32(pw + 3), gh_buf, B_, G3_, H_);
      L(k_gru_fuse, (size_t)B_ * H_,
        (const float*)(gi_big + (size_t)t * B_ * G3_), (const float*)gh_buf,
        (const float*)hEnc, hEnc, hEncB, seqh + (size_t)t * B_ * H_, B_, H_);
    }
    L(k_cvt_bf16, (size_t)S_ * B_ * H_, (const float*)seqh, seqh_bf);
    GEMM(seqh_bf, outWB, outBias, seq_out, S_ * B_, 64, H_);
    L(k_logsoftmax_time, (size_t)B_ * decDx[d], (const float*)seq_out,
      out + decOff[d], B_, S_, decDx[d], 64);
  }

  // ---------------- chroma head ----------------
  L(k_pack_w, (size_t)64 * Z_, F32(P_OUTC), outcWB, CHR_, Z_, 64, Z_);
  L(k_pack_bias, (size_t)64, F32(P_OUTC + 1), outBias, CHR_, 64);
  GEMM(zb[2], outcWB, outBias, ctmp, B_, 64, Z_);
  L(k_sigmoid_slice, (size_t)B_ * CHR_, (const float*)ctmp, out + OFF_C, CHR_, 64);

  // ---------------- global autoregressive decoder ----------------
  L(k_build_z, (size_t)B_ * 414, (const float*)zf[0], F32(4),
    (const float*)zf[1], F32(5), (const float*)zf[2], F32(3), z414);
  L(k_pack_w, (size_t)B_ * 416, (const float*)z414, z416B, B_, 414, B_, 416);
  L(k_pack_w, (size_t)H_ * 416, F32(P_LIG), ligB, H_, 414, H_, 416);
  GEMM(z416B, ligB, F32(P_LIG + 1), hx0, B_, H_, 416);
  L(k_cvt_bf16, (size_t)B_ * H_, (const float*)hx0, hx0B);
  L(k_pack_w, (size_t)G3_ * 544, F32(P_CG), cgWihB, G3_, 544, G3_, 544);
  L(k_pack_w, (size_t)G3_ * H_, F32(P_CG + 1), cgWhhB, G3_, H_, G3_, H_);
  L(k_pack_w, (size_t)G3_ * H_, F32(P_CG2), cg2WihB, G3_, H_, G3_, H_);
  L(k_pack_w, (size_t)G3_ * H_, F32(P_CG2 + 1), cg2WhhB, G3_, H_, G3_, H_);
  L(k_pack_w, (size_t)192 * H_, F32(P_OUTG), outgWB, ROLL_, H_, 192, H_);
  L(k_pack_bias, (size_t)192, F32(P_OUTG + 1), outgBias, ROLL_, 192);
  L(k_init_onehot, (size_t)B_ * ROLL_, onehot);

  for (int t = 0; t < S_; ++t) {
    L(k_build_g_inp, (size_t)B_ * 544, (const float*)onehot,
      (const float*)z414, inpgB);
    GEMM(inpgB, cgWihB, F32(P_CG + 2), giG, B_, G3_, 544);
    GEMM(hx0B, cgWhhB, F32(P_CG + 3), gh_buf, B_, G3_, H_);
    L(k_gru_fuse, (size_t)B_ * H_, (const float*)giG, (const float*)gh_buf,
      (const float*)hx0, hx0, hx0B, (float*)nullptr, B_, H_);
    GEMM(hx0B, cg2WihB, F32(P_CG2 + 2), giG, B_, G3_, H_);
    const float* h1i = (t == 0) ? (const float*)hx0 : (const float*)hx1;
    const unsigned short* h1ib = (t == 0) ? hx0B : hx1B;
    GEMM(h1ib, cg2WhhB, F32(P_CG2 + 3), gh_buf, B_, G3_, H_);
    L(k_gru_fuse, (size_t)B_ * H_, (const float*)giG, (const float*)gh_buf,
      h1i, hx1, hx1B, (float*)nullptr, B_, H_);
    GEMM(hx1B, outgWB, outgBias, logitsP, B_, 192, H_);
    L(k_lsm_argmax, (size_t)B_, (const float*)logitsP, out + OFF_OUTG, t,
      onehot);
  }
}
